// DeepAR_13048110645729
// MI455X (gfx1250) — matrix operand order repaired and checked by reading, not compiled, not run on hardware
//
#include <hip/hip_runtime.h>
#include <hip/hip_bf16.h>

// DeepAR encoder-decoder LSTM on MI455X (gfx1250, wave32, WMMA).
// B=1024, T=168, D=32, H=512, K=8, TAU=24, L=2.
//
// One bf16-WMMA GEMM + fused LSTM-pointwise kernel per (timestep, layer).
// Block tile 128M x 128N, 8 wave32s (4Mx2N), wave tile 32x64 = 2x4 WMMA frags.
// Double-buffered LDS tiles staged with GLOBAL_LOAD_ASYNC_TO_LDS_B128
// (ASYNCcnt) so staging of chunk k+1 overlaps the WMMA stream on chunk k.
// Weights are bf16, gate-interleaved columns, L2-resident across all steps.

#define Bsz  1024
#define Tenc 168
#define Din0 32
#define Hdim 512
#define Kout 8
#define TAUd 24

// Flip to 0 if the toolchain's assembler rejects the async-LDS mnemonics.
#define USE_ASYNC_STAGE 1

typedef __attribute__((ext_vector_type(16))) __bf16 v16bf;
typedef __attribute__((ext_vector_type(8)))  float  v8f;

__device__ __forceinline__ unsigned short f2bf(float f) {
  unsigned u = __float_as_uint(f);
  u += 0x7FFFu + ((u >> 16) & 1u);          // round-to-nearest-even
  return (unsigned short)(u >> 16);
}

union FragU { uint4 q[2]; v16bf v; };

// A fragment: 16x32 bf16. Lanes 0-15 -> K 0..7 & 16..23, lanes 16-31 ->
// K 8..15 & 24..31; M = lane&15. Source: row-major [16][32] bf16 (64B rows).
__device__ __forceinline__ v16bf load_a_frag(const char* rowBase, int lane) {
  int m  = lane & 15;
  int kb = (lane >> 4) << 3;                // 0 or 8
  FragU u;
  u.q[0] = *(const uint4*)(rowBase + m * 64 + kb * 2);
  u.q[1] = *(const uint4*)(rowBase + m * 64 + (16 + kb) * 2);
  return u.v;
}

// B fragment: 32x16 bf16. Lanes 0-15 -> K 0..7 & 16..23, lanes 16-31 ->
// K 8..15 & 24..31; N = lane&15. Source n-major [16][32] bf16 -> two 16B reads.
__device__ __forceinline__ v16bf load_b_frag(const char* colBase, int lane) {
  int n   = lane & 15;
  int kb2 = (lane >> 4) << 3;               // 0 or 8
  FragU u;
  const uint4* p = (const uint4*)(colBase + n * 64 + kb2 * 2);
  u.q[0] = p[0];
  u.q[1] = p[2];
  return u.v;
}

#if USE_ASYNC_STAGE
// Async direct memory->LDS (CDNA5 TDM-adjacent path, tracked by ASYNCcnt).
// ldsOff = wave-relative LDS byte offset (flat->LDS aperture keeps addr[31:0]).
__device__ __forceinline__ void async_ld_b128(unsigned ldsOff, const void* g) {
  asm volatile("global_load_async_to_lds_b128 %0, %1, off"
               :: "v"(ldsOff), "v"((unsigned long long)(uintptr_t)g)
               : "memory");
}
__device__ __forceinline__ void wait_async0() {
  asm volatile("s_wait_asynccnt 0x0" ::: "memory");
}
#endif

// ---------------------------------------------------------------------------
// Fused LSTM step: gates = [A0 | h] @ Wt + bias ; c,h pointwise update.
// ---------------------------------------------------------------------------
__global__ __launch_bounds__(256) void lstm_step_kernel(
    const unsigned short* __restrict__ A0, int K0,   // [B][K0] bf16 (x_t or h_top)
    const unsigned short* __restrict__ A1,           // [B][512] bf16 (own h_prev)
    const unsigned short* __restrict__ Wt,           // [2048][K0+512] bf16, n = 4*unit+gate
    const float* __restrict__ biasCat,               // [2048] fp32 (bih+bhh, interleaved)
    float* __restrict__ cState,                      // [B][512] fp32, in place
    unsigned short* __restrict__ hOut,               // [B][512] bf16
    unsigned short* __restrict__ seqOut,             // [B][TAU][512] bf16 or null
    int tIdx)
{
  constexpr int BM = 128, BN = 128;
  const int K1   = Hdim;
  const int Ktot = K0 + K1;
  const int tid  = threadIdx.x;
  const int lane = tid & 31;
  const int wid  = tid >> 5;
  const int waveM = (wid & 3) * 32;        // 4 waves along M
  const int waveN = (wid >> 2) * 64;       // 2 waves along N
  const int mBase = blockIdx.x * BM;
  const int nBase = blockIdx.y * BN;

  // 64KB: double-buffered A(2x8K)+B(2x8K) tiles, reused as fp32 gate tile.
  __shared__ char smem[BM * BN * 4];
  // tile offsets within smem (bytes)
  //   A buf p: p*8192          [128 rows][32 k] bf16
  //   B buf p: 16384 + p*8192  [128 n  ][32 k] bf16

  v8f zero8 = {0.f,0.f,0.f,0.f,0.f,0.f,0.f,0.f};
  v8f acc[2][4];
  for (int i = 0; i < 2; ++i)
    for (int j = 0; j < 4; ++j) acc[i][j] = zero8;

  const int sRow  = tid >> 1;              // 0..127
  const int sHalf = tid & 1;               // two 32B halves per 64B row
#if USE_ASYNC_STAGE
  const unsigned smemBase = (unsigned)(uintptr_t)smem;
#endif

  // stage one 32-chunk of A and B into LDS buffer `buf`
  auto stage = [&](int kb, int buf) {
    const unsigned short* src; int srcK, kOff;
    if (kb < K0) { src = A0; srcK = K0; kOff = kb; }
    else         { src = A1; srcK = K1; kOff = kb - K0; }
    const unsigned short* ga = src + (size_t)(mBase + sRow) * srcK + kOff + sHalf * 16;
    const unsigned short* gb = Wt + (size_t)(nBase + sRow) * Ktot + kb + sHalf * 16;
#if USE_ASYNC_STAGE
    unsigned da = smemBase + buf * 8192 + sRow * 64 + sHalf * 32;
    unsigned db = smemBase + 16384 + buf * 8192 + sRow * 64 + sHalf * 32;
    async_ld_b128(da,      ga);
    async_ld_b128(da + 16, ga + 8);
    async_ld_b128(db,      gb);
    async_ld_b128(db + 16, gb + 8);
#else
    uint4* da = (uint4*)(smem + buf * 8192 + sRow * 64 + sHalf * 32);
    uint4* db = (uint4*)(smem + 16384 + buf * 8192 + sRow * 64 + sHalf * 32);
    const uint4* pa = (const uint4*)ga;
    const uint4* pb = (const uint4*)gb;
    da[0] = pa[0]; da[1] = pa[1];
    db[0] = pb[0]; db[1] = pb[1];
#endif
    __builtin_prefetch(gb + 32, 0, 0);     // global_prefetch_b8: next weight chunk
  };

  const int NC = Ktot / 32;
  stage(0, 0);
#if USE_ASYNC_STAGE
  wait_async0();
#endif
  __syncthreads();

  for (int kc = 0; kc < NC; ++kc) {
    const int cur = kc & 1;
    if (kc + 1 < NC) stage((kc + 1) * 32, cur ^ 1);   // overlap with WMMA below

    const char* Ab = smem + cur * 8192;
    const char* Bb = smem + 16384 + cur * 8192;
    v16bf a0 = load_a_frag(Ab + (waveM + 0)  * 64, lane);
    v16bf a1 = load_a_frag(Ab + (waveM + 16) * 64, lane);
    v16bf b0 = load_b_frag(Bb + (waveN + 0)  * 64, lane);
    v16bf b1 = load_b_frag(Bb + (waveN + 16) * 64, lane);
    v16bf b2 = load_b_frag(Bb + (waveN + 32) * 64, lane);
    v16bf b3 = load_b_frag(Bb + (waveN + 48) * 64, lane);
    acc[0][0] = __builtin_amdgcn_wmma_f32_16x16x32_bf16(false, a0, false, b0, (short)0, acc[0][0], false, false);
    acc[0][1] = __builtin_amdgcn_wmma_f32_16x16x32_bf16(false, a0, false, b1, (short)0, acc[0][1], false, false);
    acc[0][2] = __builtin_amdgcn_wmma_f32_16x16x32_bf16(false, a0, false, b2, (short)0, acc[0][2], false, false);
    acc[0][3] = __builtin_amdgcn_wmma_f32_16x16x32_bf16(false, a0, false, b3, (short)0, acc[0][3], false, false);
    acc[1][0] = __builtin_amdgcn_wmma_f32_16x16x32_bf16(false, a1, false, b0, (short)0, acc[1][0], false, false);
    acc[1][1] = __builtin_amdgcn_wmma_f32_16x16x32_bf16(false, a1, false, b1, (short)0, acc[1][1], false, false);
    acc[1][2] = __builtin_amdgcn_wmma_f32_16x16x32_bf16(false, a1, false, b2, (short)0, acc[1][2], false, false);
    acc[1][3] = __builtin_amdgcn_wmma_f32_16x16x32_bf16(false, a1, false, b3, (short)0, acc[1][3], false, false);

#if USE_ASYNC_STAGE
    if (kc + 1 < NC) wait_async0();        // next-chunk staging landed in LDS
#endif
    __syncthreads();                        // + all ds reads of `cur` retired
  }

  // -- stage gates (fp32 + bias) to LDS; C layout: M = vgpr + 8*(lane>=16) --
  float* gates = (float*)smem;             // [128][128]
  const int ml = (lane >> 4) * 8;
  const int nl = lane & 15;
  for (int i = 0; i < 2; ++i)
    for (int j = 0; j < 4; ++j)
      for (int e = 0; e < 8; ++e) {
        int m = waveM + i * 16 + ml + e;
        int n = waveN + j * 16 + nl;
        gates[m * BN + n] = acc[i][j][e] + biasCat[nBase + n];
      }
  __syncthreads();

  // -- LSTM pointwise: this tile owns all 4 gates of 32 hidden units --
  for (int it = tid; it < BM * 32; it += 256) {
    int m = it >> 5, j = it & 31;
    const float* g4 = gates + m * BN + j * 4;
    float gi = g4[0], gf = g4[1], gg = g4[2], go = g4[3];
    int b   = mBase + m;
    int col = (nBase >> 2) + j;
    size_t idx = (size_t)b * Hdim + col;
    float cv = cState[idx];
    float si = 1.f / (1.f + __expf(-gi));
    float sf = 1.f / (1.f + __expf(-gf));
    float so = 1.f / (1.f + __expf(-go));
    float cn = sf * cv + si * tanhf(gg);
    float hn = so * tanhf(cn);
    cState[idx] = cn;
    unsigned short hb = f2bf(hn);
    hOut[idx] = hb;
    if (seqOut) seqOut[((size_t)b * TAUd + tIdx) * Hdim + col] = hb;
  }
}

// ---------------------------------------------------------------------------
// Output heads: mu/sigma packed as one [24576 x 512] x [512 x 16] WMMA GEMM
// (cols 0-7 = W1/mu, cols 8-15 = W2/sigma-preact), softplus fused.
// ---------------------------------------------------------------------------
__global__ __launch_bounds__(256) void head_kernel(
    const unsigned short* __restrict__ seq,   // [B*TAU][512] bf16
    const unsigned short* __restrict__ Wh,    // [16][512] bf16
    const float* __restrict__ b1,
    const float* __restrict__ b2,
    float* __restrict__ mu,
    float* __restrict__ sigma)
{
  const int lane  = threadIdx.x & 31;
  const int wid   = threadIdx.x >> 5;
  const int row0  = blockIdx.x * 128 + wid * 16;
  const int mlane = lane & 15;
  const int kbA   = (lane >> 4) << 3;
  const int kbB   = (lane >> 4) << 3;

  const char* aRow = (const char*)(seq + (size_t)(row0 + mlane) * Hdim);
  const char* bRow = (const char*)(Wh + (size_t)mlane * Hdim);   // n = mlane row

  v8f acc = {0.f,0.f,0.f,0.f,0.f,0.f,0.f,0.f};
  for (int k = 0; k < Hdim; k += 32) {
    FragU ua, ub;
    ua.q[0] = *(const uint4*)(aRow + (k + kbA) * 2);
    ua.q[1] = *(const uint4*)(aRow + (k + 16 + kbA) * 2);
    const uint4* bp = (const uint4*)(bRow + (k + kbB) * 2);
    ub.q[0] = bp[0]; ub.q[1] = bp[2];
    acc = __builtin_amdgcn_wmma_f32_16x16x32_bf16(false, ua.v, false, ub.v, (short)0, acc, false, false);
  }
  const int ml = (lane >> 4) * 8;
  const int nl = lane & 15;
  for (int e = 0; e < 8; ++e) {
    int r = row0 + ml + e;
    float v = acc[e];
    if (nl < 8) {
      mu[(size_t)r * Kout + nl] = v + b1[nl];
    } else {
      float t  = 2.f * (v + b2[nl - 8]);                 // beta = 2
      float sp = (t > 20.f) ? t : log1pf(__expf(t));     // stable softplus
      sigma[(size_t)r * Kout + (nl - 8)] = 0.5f * sp;
    }
  }
}

// ---------------------------------------------------------------------------
// One-time-per-launch conversions (deterministic; re-run every call).
// ---------------------------------------------------------------------------
__global__ void conv_w_kernel(const float* __restrict__ Wih, const float* __restrict__ Whh,
                              const float* __restrict__ bih, const float* __restrict__ bhh,
                              int Din, unsigned short* __restrict__ Wt,
                              float* __restrict__ biasCat)
{
  const int n = blockIdx.x;            // gate-interleaved column 0..2047
  const int j = n >> 2, g = n & 3;     // unit, gate (PyTorch order i,f,g,o)
  const int r = g * Hdim + j;          // original row in [4H]
  const int Ktot = Din + Hdim;
  for (int k = threadIdx.x; k < Ktot; k += blockDim.x) {
    float v = (k < Din) ? Wih[(size_t)r * Din + k] : Whh[(size_t)r * Hdim + (k - Din)];
    Wt[(size_t)n * Ktot + k] = f2bf(v);
  }
  if (threadIdx.x == 0) biasCat[n] = bih[r] + bhh[r];
}

__global__ void conv_x_kernel(const float* __restrict__ x, unsigned short* __restrict__ xT)
{
  size_t idx = (size_t)blockIdx.x * blockDim.x + threadIdx.x;
  const size_t total = (size_t)Bsz * Tenc * Din0;
  if (idx >= total) return;
  int d = (int)(idx % Din0);
  int t = (int)((idx / Din0) % Tenc);
  int b = (int)(idx / ((size_t)Din0 * Tenc));
  xT[((size_t)t * Bsz + b) * Din0 + d] = f2bf(x[idx]);   // [T][B][D]
}

__global__ void conv_head_kernel(const float* __restrict__ W1, const float* __restrict__ W2,
                                 unsigned short* __restrict__ Wh)
{
  int idx = blockIdx.x * blockDim.x + threadIdx.x;
  if (idx >= 16 * Hdim) return;
  int n = idx / Hdim, k = idx % Hdim;
  float v = (n < 8) ? W1[(size_t)n * Hdim + k] : W2[(size_t)(n - 8) * Hdim + k];
  Wh[idx] = f2bf(v);
}

__global__ void zero_kernel(uint4* __restrict__ p, size_t n16)
{
  size_t i = (size_t)blockIdx.x * blockDim.x + threadIdx.x;
  if (i < n16) p[i] = make_uint4(0u, 0u, 0u, 0u);
}

// ---------------------------------------------------------------------------
extern "C" void kernel_launch(void* const* d_in, const int* in_sizes, int n_in,
                              void* d_out, int out_size, void* d_ws, size_t ws_size,
                              hipStream_t stream)
{
  (void)in_sizes; (void)n_in; (void)out_size; (void)ws_size;
  const float* x     = (const float*)d_in[0];
  const float* eWih0 = (const float*)d_in[1];  const float* eWhh0 = (const float*)d_in[2];
  const float* ebih0 = (const float*)d_in[3];  const float* ebhh0 = (const float*)d_in[4];
  const float* eWih1 = (const float*)d_in[5];  const float* eWhh1 = (const float*)d_in[6];
  const float* ebih1 = (const float*)d_in[7];  const float* ebhh1 = (const float*)d_in[8];
  const float* dWih0 = (const float*)d_in[9];  const float* dWhh0 = (const float*)d_in[10];
  const float* dbih0 = (const float*)d_in[11]; const float* dbhh0 = (const float*)d_in[12];
  const float* dWih1 = (const float*)d_in[13]; const float* dWhh1 = (const float*)d_in[14];
  const float* dbih1 = (const float*)d_in[15]; const float* dbhh1 = (const float*)d_in[16];
  const float* W1 = (const float*)d_in[17];    const float* b1 = (const float*)d_in[18];
  const float* W2 = (const float*)d_in[19];    const float* b2 = (const float*)d_in[20];

  // workspace carve (~60 MB), 256B-aligned regions
  char* ws = (char*)d_ws;
  auto carve = [&](size_t bytes) -> char* {
    char* p = ws; ws += (bytes + 255) & ~(size_t)255; return p;
  };
  unsigned short* wtE0 = (unsigned short*)carve((size_t)2048 * 544  * 2);
  unsigned short* wtE1 = (unsigned short*)carve((size_t)2048 * 1024 * 2);
  unsigned short* wtD0 = (unsigned short*)carve((size_t)2048 * 1024 * 2);
  unsigned short* wtD1 = (unsigned short*)carve((size_t)2048 * 1024 * 2);
  float* biasE0 = (float*)carve(2048 * 4);
  float* biasE1 = (float*)carve(2048 * 4);
  float* biasD0 = (float*)carve(2048 * 4);
  float* biasD1 = (float*)carve(2048 * 4);
  unsigned short* Wh  = (unsigned short*)carve((size_t)16 * Hdim * 2);
  unsigned short* xT  = (unsigned short*)carve((size_t)Tenc * Bsz * Din0 * 2);
  unsigned short* hBuf = (unsigned short*)carve((size_t)2 * 2 * Bsz * Hdim * 2);
  float*          cBuf = (float*)carve((size_t)2 * Bsz * Hdim * 4);
  unsigned short* seq  = (unsigned short*)carve((size_t)Bsz * TAUd * Hdim * 2);

  const size_t BH = (size_t)Bsz * Hdim;
  unsigned short* h0[2] = { hBuf,          hBuf + BH     };
  unsigned short* h1[2] = { hBuf + 2 * BH, hBuf + 3 * BH };
  float* c0 = cBuf;
  float* c1 = cBuf + BH;

  // zero h (both ping-pong bufs, both layers) and c — contiguous regions
  {
    size_t n16 = (4 * BH * 2 /*h bytes*/ + 2 * BH * 4 /*c bytes*/) / 16;
    zero_kernel<<<(unsigned)((n16 + 255) / 256), 256, 0, stream>>>((uint4*)hBuf, n16);
  }
  conv_w_kernel<<<2048, 256, 0, stream>>>(eWih0, eWhh0, ebih0, ebhh0, Din0, wtE0, biasE0);
  conv_w_kernel<<<2048, 256, 0, stream>>>(eWih1, eWhh1, ebih1, ebhh1, Hdim, wtE1, biasE1);
  conv_w_kernel<<<2048, 256, 0, stream>>>(dWih0, dWhh0, dbih0, dbhh0, Hdim, wtD0, biasD0);
  conv_w_kernel<<<2048, 256, 0, stream>>>(dWih1, dWhh1, dbih1, dbhh1, Hdim, wtD1, biasD1);
  {
    size_t total = (size_t)Bsz * Tenc * Din0;
    conv_x_kernel<<<(unsigned)((total + 255) / 256), 256, 0, stream>>>(x, xT);
  }
  conv_head_kernel<<<(16 * Hdim + 255) / 256, 256, 0, stream>>>(W1, W2, Wh);

  const dim3 grid(Bsz / 128, 2048 / 128);   // (8, 16) = 128 blocks per step-layer

  // encoder: layer0 input = x_t (K0=32), layer1 input = h0_new
  for (int s = 0; s < Tenc; ++s) {
    int rp = s & 1, wp = rp ^ 1;
    lstm_step_kernel<<<grid, 256, 0, stream>>>(xT + (size_t)s * Bsz * Din0, Din0,
                                               h0[rp], wtE0, biasE0, c0, h0[wp], nullptr, 0);
    lstm_step_kernel<<<grid, 256, 0, stream>>>(h0[wp], Hdim,
                                               h1[rp], wtE1, biasE1, c1, h1[wp], nullptr, 0);
  }
  // decoder: layer0 input = previous top hidden; top output streamed to seq
  for (int td = 0; td < TAUd; ++td) {
    int s = Tenc + td;
    int rp = s & 1, wp = rp ^ 1;
    lstm_step_kernel<<<grid, 256, 0, stream>>>(h1[rp], Hdim,
                                               h0[rp], wtD0, biasD0, c0, h0[wp], nullptr, 0);
    lstm_step_kernel<<<grid, 256, 0, stream>>>(h0[wp], Hdim,
                                               h1[rp], wtD1, biasD1, c1, h1[wp], seq, td);
  }

  float* mu    = (float*)d_out;                       // [B,TAU,8]
  float* sigma = mu + (size_t)Bsz * TAUd * Kout;      // [B,TAU,8]
  head_kernel<<<(Bsz * TAUd) / 128, 256, 0, stream>>>(seq, Wh, b1, b2, mu, sigma);
}
